// VisionMambaBlock_61718680044223
// MI455X (gfx1250) — compile-verified, matrix operand order checked by reading, not run on hardware
//
#include <hip/hip_runtime.h>
#include <stdint.h>

typedef __attribute__((ext_vector_type(16))) __bf16 v16bf;
typedef __attribute__((ext_vector_type(8)))  float  v8f;

#define DIMC   192
#define DSTATE 16
#define DCONV  4
#define DI     384
#define DTR    12
#define NB     2
#define LSEQ   4096
#define MROWS  (NB * LSEQ)        // 8192 rows (B*L)
#define NXZ    (2 * DI)           // 768
#define NDBC   (DTR + 2 * DSTATE) // 44

__device__ __forceinline__ uint16_t f2bf(float f) {
  uint32_t u = __float_as_uint(f);
  u += 0x7FFFu + ((u >> 16) & 1u);   // round-to-nearest-even
  return (uint16_t)(u >> 16);
}

// ---------------- fp32 -> bf16 weight conversion ----------------
__global__ void k_f32_to_bf16(const float* __restrict__ src,
                              uint16_t* __restrict__ dst, int n) {
  int i = blockIdx.x * blockDim.x + threadIdx.x;
  if (i < n) dst[i] = f2bf(src[i]);
}

// ---------------- LayerNorm: NCHW x -> (B*L, C) xn (f32 + bf16) ----------------
__global__ void k_layernorm(const float* __restrict__ x, const float* __restrict__ g,
                            const float* __restrict__ b, float* __restrict__ xn,
                            uint16_t* __restrict__ xnb) {
  __shared__ float s1[256], s2[256];
  int row = blockIdx.x;                 // bb*L + t
  int bb = row / LSEQ, t = row % LSEQ;
  int tid = threadIdx.x;
  float v = 0.f;
  if (tid < DIMC) v = x[(size_t)bb * DIMC * LSEQ + (size_t)tid * LSEQ + t];
  s1[tid] = (tid < DIMC) ? v : 0.f;
  s2[tid] = (tid < DIMC) ? v * v : 0.f;
  __syncthreads();
  for (int o = 128; o > 0; o >>= 1) {
    if (tid < o) { s1[tid] += s1[tid + o]; s2[tid] += s2[tid + o]; }
    __syncthreads();
  }
  float mean = s1[0] / DIMC;
  float var  = s2[0] / DIMC - mean * mean;
  float inv  = rsqrtf(var + 1e-5f);
  if (tid < DIMC) {
    float o = (v - mean) * inv * g[tid] + b[tid];
    xn[(size_t)row * DIMC + tid]  = o;
    xnb[(size_t)row * DIMC + tid] = f2bf(o);
  }
}

// ---------------- WMMA bf16 GEMM: C[M,N] = A[M,K] * W[N,K]^T ----------------
// One 16x16 tile per wave, K in steps of 32, v_wmma_f32_16x16x32_bf16.
union ABfrag { uint4 u[2]; v16bf v; };

__global__ void k_wmma_gemm(const uint16_t* __restrict__ A, const uint16_t* __restrict__ W,
                            float* __restrict__ C, int M, int N, int K, int flip) {
  int lane  = threadIdx.x & 31;
  int wave  = threadIdx.x >> 5;
  int tileN = blockIdx.x * (blockDim.x >> 5) + wave;
  int tileM = blockIdx.y;
  int n  = tileN * 16 + (lane & 15);
  int m  = tileM * 16 + (lane & 15);
  int mA = m;
  if (flip) { int bb = m / LSEQ, t = m % LSEQ; mA = bb * LSEQ + (LSEQ - 1 - t); }
  int hiK = (lane >= 16) ? 8 : 0;       // per-ISA 16-bit A/B fragment striping
  const uint16_t* arow = A + (size_t)mA * K;
  const uint16_t* wrow = W + (size_t)n  * K;
  v8f acc = {};
  for (int k0 = 0; k0 < K; k0 += 32) {
    ABfrag a, b;
    a.u[0] = *(const uint4*)(arow + k0 + hiK);
    a.u[1] = *(const uint4*)(arow + k0 + hiK + 16);
    b.u[0] = *(const uint4*)(wrow + k0 + hiK);
    b.u[1] = *(const uint4*)(wrow + k0 + hiK + 16);
    acc = __builtin_amdgcn_wmma_f32_16x16x32_bf16(false, a.v, false, b.v,
                                                  (short)0, acc, false, false);
  }
  int mo = tileM * 16 + ((lane >= 16) ? 8 : 0);
#pragma unroll
  for (int v = 0; v < 8; ++v)
    C[(size_t)(mo + v) * N + n] = acc[v];
}

// ---------------- causal depthwise conv (k=4) + bias + silu ----------------
__global__ void k_conv_silu(const float* __restrict__ xz, const float* __restrict__ cw,
                            const float* __restrict__ cb, float* __restrict__ u, int total) {
  int i = blockIdx.x * blockDim.x + threadIdx.x;
  if (i >= total) return;
  int d = i % DI;
  int m = i / DI;                       // bb*L + t
  int t = m % LSEQ;
  float acc = cb[d];
#pragma unroll
  for (int j = 0; j < DCONV; ++j) {
    int tt = t - (DCONV - 1) + j;
    if (tt >= 0)
      acc += xz[(size_t)(m - (DCONV - 1) + j) * NXZ + d] * cw[d * DCONV + j];
  }
  u[(size_t)m * DI + d] = acc / (1.f + __expf(-acc));   // silu
}

// ---------------- dbc[m, 44] = u[m,:] . xpW[j,:]  (small N, scalar) ----------------
__global__ void k_dbc(const float* __restrict__ u, const float* __restrict__ xpW,
                      float* __restrict__ dbc, int total) {
  int i = blockIdx.x * blockDim.x + threadIdx.x;
  if (i >= total) return;
  int j = i % NDBC;
  int m = i / NDBC;
  const float* ur = u + (size_t)m * DI;
  const float* wr = xpW + (size_t)j * DI;
  float acc = 0.f;
#pragma unroll 4
  for (int k = 0; k < DI; ++k) acc += ur[k] * wr[k];
  dbc[(size_t)m * NDBC + j] = acc;
}

// ---------------- delta = softplus(dt @ dtW^T + dtB), K=12 ----------------
__global__ void k_delta(const float* __restrict__ dbc, const float* __restrict__ dtW,
                        const float* __restrict__ dtB, float* __restrict__ delta, int total) {
  int i = blockIdx.x * blockDim.x + threadIdx.x;
  if (i >= total) return;
  int d = i % DI;
  int m = i / DI;
  float acc = dtB[d];
  const float* r = dbc + (size_t)m * NDBC;
#pragma unroll
  for (int k = 0; k < DTR; ++k) acc += r[k] * dtW[d * DTR + k];
  delta[(size_t)m * DI + d] = (acc > 20.f) ? acc : log1pf(__expf(acc));
}

// ---------------- selective scan + D skip + silu(z) gate -> g (bf16) ----------------
// lane = one (channel, state); 16 lanes per channel; 2 channels per wave32.
__global__ void k_scan(const float* __restrict__ delta, const float* __restrict__ u,
                       const float* __restrict__ dbc, const float* __restrict__ xz,
                       const float* __restrict__ Alog, const float* __restrict__ Dp,
                       uint16_t* __restrict__ g) {
  int lane = threadIdx.x & 31;
  int wave = (blockIdx.x * blockDim.x + threadIdx.x) >> 5;
  int ch = wave * 2 + (lane >> 4);      // 0 .. B*DI-1
  int n  = lane & 15;                   // state index
  int bb = ch / DI;
  int d  = ch % DI;
  float A  = -__expf(Alog[d * DSTATE + n]);
  float Dd = Dp[d];
  float h  = 0.f;
  size_t base = (size_t)bb * LSEQ;
  for (int t = 0; t < LSEQ; ++t) {
    size_t m = base + t;
    float dt = delta[m * DI + d];
    float ut = u[m * DI + d];
    float Bt = dbc[m * NDBC + DTR + n];
    float Ct = dbc[m * NDBC + DTR + DSTATE + n];
    h = __expf(dt * A) * h + dt * ut * Bt;
    float v = h * Ct;
#pragma unroll
    for (int o = 8; o > 0; o >>= 1) v += __shfl_xor(v, o, 16);
    if (n == 0) {
      float z = xz[m * NXZ + DI + d];
      float y = v + ut * Dd;
      g[m * DI + d] = f2bf(y * (z / (1.f + __expf(-z))));
    }
  }
}

// ---------------- comb = of + flip_L(ob), bf16 for proj GEMM ----------------
__global__ void k_combine(const float* __restrict__ of, const float* __restrict__ ob,
                          uint16_t* __restrict__ comb, int total) {
  int i = blockIdx.x * blockDim.x + threadIdx.x;
  if (i >= total) return;
  int c = i % DIMC;
  int m = i / DIMC;
  int bb = m / LSEQ, t = m % LSEQ;
  float v = of[(size_t)m * DIMC + c] +
            ob[((size_t)bb * LSEQ + (LSEQ - 1 - t)) * DIMC + c];
  comb[i] = f2bf(v);
}

// ---------------- residual add + bias, back to NCHW ----------------
__global__ void k_final(const float* __restrict__ x, const float* __restrict__ tmp,
                        const float* __restrict__ pb, float* __restrict__ out, int total) {
  int i = blockIdx.x * blockDim.x + threadIdx.x;
  if (i >= total) return;
  int t   = i % LSEQ;                   // H*W position
  int rem = i / LSEQ;
  int c   = rem % DIMC;
  int bb  = rem / DIMC;
  out[i] = x[i] + tmp[((size_t)bb * LSEQ + t) * DIMC + c] + pb[c];
}

extern "C" void kernel_launch(void* const* d_in, const int* in_sizes, int n_in,
                              void* d_out, int out_size, void* d_ws, size_t ws_size,
                              hipStream_t stream) {
  const float* x     = (const float*)d_in[0];
  const float* ln_g  = (const float*)d_in[1];
  const float* ln_b  = (const float*)d_in[2];
  const float* projW = (const float*)d_in[3];
  const float* projb = (const float*)d_in[4];

  char* ws = (char*)d_ws;
  size_t off = 0;
  auto alloc = [&](size_t bytes) -> void* {
    void* p = ws + off;
    off = (off + bytes + 255) & ~(size_t)255;
    return p;
  };

  float*    xn    = (float*)   alloc((size_t)MROWS * DIMC * 4);
  uint16_t* xnb   = (uint16_t*)alloc((size_t)MROWS * DIMC * 2);
  float*    xz    = (float*)   alloc((size_t)MROWS * NXZ  * 4);
  float*    u     = (float*)   alloc((size_t)MROWS * DI   * 4);
  float*    dbc   = (float*)   alloc((size_t)MROWS * NDBC * 4);
  float*    delta = (float*)   alloc((size_t)MROWS * DI   * 4);
  uint16_t* gbuf  = (uint16_t*)alloc((size_t)MROWS * DI   * 2);
  float*    odir0 = (float*)   alloc((size_t)MROWS * DIMC * 4);
  float*    odir1 = (float*)   alloc((size_t)MROWS * DIMC * 4);
  uint16_t* inWb  = (uint16_t*)alloc((size_t)NXZ  * DIMC * 2);
  uint16_t* outWb = (uint16_t*)alloc((size_t)DIMC * DI   * 2);
  uint16_t* projWb= (uint16_t*)alloc((size_t)DIMC * DIMC * 2);
  uint16_t* comb  = gbuf;   // reuse: scan output no longer needed at combine
  float*    tmp   = xz;     // reuse: xz no longer needed after last scan

  // LayerNorm + projW conversion
  k_layernorm<<<MROWS, 256, 0, stream>>>(x, ln_g, ln_b, xn, xnb);
  {
    int npw = DIMC * DIMC;
    k_f32_to_bf16<<<(npw + 255) / 256, 256, 0, stream>>>(projW, projWb, npw);
  }

  for (int dir = 0; dir < 2; ++dir) {
    int base = 5 + dir * 9;
    const float* inW   = (const float*)d_in[base + 0];
    const float* convW = (const float*)d_in[base + 1];
    const float* convB = (const float*)d_in[base + 2];
    const float* xpW   = (const float*)d_in[base + 3];
    const float* dtW   = (const float*)d_in[base + 4];
    const float* dtB   = (const float*)d_in[base + 5];
    const float* Alog  = (const float*)d_in[base + 6];
    const float* Dp    = (const float*)d_in[base + 7];
    const float* outW  = (const float*)d_in[base + 8];

    int niw = NXZ * DIMC;
    k_f32_to_bf16<<<(niw + 255) / 256, 256, 0, stream>>>(inW, inWb, niw);
    int now = DIMC * DI;
    k_f32_to_bf16<<<(now + 255) / 256, 256, 0, stream>>>(outW, outWb, now);

    // xz = xn @ inW^T   (M=8192, N=768, K=192); dir b reads xn with flipped L
    dim3 g1(NXZ / 64, MROWS / 16);
    k_wmma_gemm<<<g1, 128, 0, stream>>>(xnb, inWb, xz, MROWS, NXZ, DIMC, dir);

    int tot_u = MROWS * DI;
    k_conv_silu<<<(tot_u + 255) / 256, 256, 0, stream>>>(xz, convW, convB, u, tot_u);

    int tot_dbc = MROWS * NDBC;
    k_dbc<<<(tot_dbc + 255) / 256, 256, 0, stream>>>(u, xpW, dbc, tot_dbc);

    k_delta<<<(tot_u + 255) / 256, 256, 0, stream>>>(dbc, dtW, dtB, delta, tot_u);

    // 768 channels, 2 per wave -> 384 waves -> 48 blocks x 256 threads
    k_scan<<<48, 256, 0, stream>>>(delta, u, dbc, xz, Alog, Dp, gbuf);

    // odir = g @ outW^T  (M=8192, N=192, K=384)
    dim3 g2(DIMC / 64, MROWS / 16);
    k_wmma_gemm<<<g2, 128, 0, stream>>>(gbuf, outWb, dir ? odir1 : odir0,
                                        MROWS, DIMC, DI, 0);
  }

  int tot_c = MROWS * DIMC;
  k_combine<<<(tot_c + 255) / 256, 256, 0, stream>>>(odir0, odir1, comb, tot_c);

  // tmp = comb @ projW^T  (M=8192, N=192, K=192)
  dim3 g3(DIMC / 64, MROWS / 16);
  k_wmma_gemm<<<g3, 128, 0, stream>>>(comb, projWb, tmp, MROWS, DIMC, DIMC, 0);

  k_final<<<(tot_c + 255) / 256, 256, 0, stream>>>(x, tmp, projb, (float*)d_out, tot_c);
}
